// MemoryTransformer_70231305224456
// MI455X (gfx1250) — compile-verified, matrix operand order checked by reading, not run on hardware
//
#include <hip/hip_runtime.h>
#include <hip/hip_bf16.h>

typedef __attribute__((ext_vector_type(16))) __bf16 v16bf;
typedef __attribute__((ext_vector_type(8)))  __bf16 v8bf;
typedef __attribute__((ext_vector_type(8)))  float  v8f;

#define SEQ   512
#define BATCH 8
#define DIM   512
#define HEADS 8
#define DH    64
#define FFD   2048
#define MEMN  4096
#define TOPK  32
#define DEPTH 6

// ---------------------------------------------------------------------------
// Generic strided-batched bf16 WMMA GEMM:  C = alpha * A x B (+ resid)
// A: [M,K] bf16 row-major.  B: [K,N] row-major (TRANSB=0) or [N,K] (TRANSB=1).
// Batch z decomposed per tensor (power-of-two inner counts, shift/mask):
//   off = (z >> shift)*sOuter + (z & mask)*sInner.
// zBase shifts the z used for A/B (chunked batches); C/resid use local z.
// Two-stage software pipeline: double-buffered LDS, one barrier per k-tile.
// ---------------------------------------------------------------------------
struct GemmArgs {
  const __bf16* A; const __bf16* B;
  float* Cf; __bf16* Cb; const float* R;
  int M, N, K, lda, ldb, ldc;
  int zShiftA, zShiftB, zShiftC, zBase;
  long long sAo, sAi, sBo, sBi, sCo, sCi;
  float alpha;
};

template <int TRANSB>
__global__ __launch_bounds__(256)
void gemm_bf16_wmma(GemmArgs g) {
  const int z  = blockIdx.z;
  const int zz = z + g.zBase;
  const __bf16* A = g.A + (long long)(zz >> g.zShiftA) * g.sAo
                        + (long long)(zz & ((1 << g.zShiftA) - 1)) * g.sAi;
  const __bf16* B = g.B + (long long)(zz >> g.zShiftB) * g.sBo
                        + (long long)(zz & ((1 << g.zShiftB) - 1)) * g.sBi;
  const long long cOff = (long long)(z >> g.zShiftC) * g.sCo
                       + (long long)(z & ((1 << g.zShiftC) - 1)) * g.sCi;

  const int m0 = blockIdx.y * 128;
  const int n0 = blockIdx.x * 128;
  const int tid  = threadIdx.x;
  const int lane = tid & 31;
  const int wave = tid >> 5;
  const int wm = (wave >> 2) * 64;   // 2 waves along M
  const int wn = (wave & 3) * 32;    // 4 waves along N

  __shared__ __align__(16) __bf16 As[2][128 * 40];
  __shared__ __align__(16) __bf16 Bs[2][128 * 40];   // N-major: Bs[n][k]

  v8f acc[4][2] = {};

  // staging indices (loop-invariant)
  const int arow = tid >> 1, acol = (tid & 1) * 16;
  const int btn  = tid >> 1, btk  = (tid & 1) * 16;   // TRANSB=1 path
  const int bk   = tid >> 3, bnb  = (tid & 7) * 16;   // TRANSB=0 path
  const bool bFull = TRANSB ? true : (n0 + bnb + 16 <= g.N);

  v8bf ra0, ra1, rb0, rb1;   // staged global data for the next tile

  auto globalLoad = [&](int k0) {
    const __bf16* srcA = A + (long long)(m0 + arow) * g.lda + (k0 + acol);
    ra0 = *(const v8bf*)srcA;
    ra1 = *(const v8bf*)(srcA + 8);
    if (TRANSB) {
      const __bf16* srcB = B + (long long)(n0 + btn) * g.ldb + (k0 + btk);
      rb0 = *(const v8bf*)srcB;
      rb1 = *(const v8bf*)(srcB + 8);
    } else {
      const __bf16* srcB = B + (long long)(k0 + bk) * g.ldb + (n0 + bnb);
      if (bFull) {
        rb0 = *(const v8bf*)srcB;
        rb1 = *(const v8bf*)(srcB + 8);
      } else {
#pragma unroll
        for (int i = 0; i < 8; ++i) {
          rb0[i] = (n0 + bnb + i     < g.N) ? srcB[i]     : (__bf16)0.0f;
          rb1[i] = (n0 + bnb + 8 + i < g.N) ? srcB[8 + i] : (__bf16)0.0f;
        }
      }
    }
  };

  auto ldsStore = [&](int buf) {
    *(v8bf*)(&As[buf][arow * 40 + acol])     = ra0;
    *(v8bf*)(&As[buf][arow * 40 + acol + 8]) = ra1;
    if (TRANSB) {
      *(v8bf*)(&Bs[buf][btn * 40 + btk])     = rb0;
      *(v8bf*)(&Bs[buf][btn * 40 + btk + 8]) = rb1;
    } else {
#pragma unroll
      for (int i = 0; i < 8; ++i) {
        Bs[buf][(bnb + i) * 40 + bk]     = rb0[i];
        Bs[buf][(bnb + 8 + i) * 40 + bk] = rb1[i];
      }
    }
  };

  // Fragment layout per ISA 7.12.2 (16-bit A 16x32): lane = M (or N) index,
  // kbase = (lane>>4)*8, 8-elem halves at +0 and +16.
  const int fr  = lane & 15;
  const int kb2 = (lane >> 4) * 8;

  auto compute = [&](int buf) {
    v16bf af[4], bfv[2];
#pragma unroll
    for (int tm = 0; tm < 4; ++tm) {
      const __bf16* p = &As[buf][(wm + tm * 16 + fr) * 40];
      union { v16bf v; v8bf h[2]; } u;
      u.h[0] = *(const v8bf*)(p + kb2);
      u.h[1] = *(const v8bf*)(p + kb2 + 16);
      af[tm] = u.v;
    }
#pragma unroll
    for (int tn = 0; tn < 2; ++tn) {
      const __bf16* p = &Bs[buf][(wn + tn * 16 + fr) * 40];
      union { v16bf v; v8bf h[2]; } u;
      u.h[0] = *(const v8bf*)(p + kb2);
      u.h[1] = *(const v8bf*)(p + kb2 + 16);
      bfv[tn] = u.v;
    }
#pragma unroll
    for (int tm = 0; tm < 4; ++tm)
#pragma unroll
      for (int tn = 0; tn < 2; ++tn)
        acc[tm][tn] = __builtin_amdgcn_wmma_f32_16x16x32_bf16(
            false, af[tm], false, bfv[tn], (short)0, acc[tm][tn], false, false);
  };

  const int nt = g.K >> 5;   // K is a multiple of 32 for every call site
  globalLoad(0);
  ldsStore(0);
  if (nt > 1) globalLoad(32);

  for (int t = 0; t < nt; ++t) {
    __syncthreads();                       // tile t visible; buf (t+1)&1 free
    if (t + 1 < nt) ldsStore((t + 1) & 1); // consume staged regs (tile t+1)
    if (t + 2 < nt) globalLoad((t + 2) << 5); // refill staged regs early
    if (t + 3 < nt)                        // L2 prefetch (global_prefetch_b8)
      __builtin_prefetch(A + (long long)(m0 + arow) * g.lda +
                             (((t + 3) << 5) + acol), 0, 1);
    compute(t & 1);
  }

  float*  Cf = g.Cf ? g.Cf + cOff : nullptr;
  __bf16* Cb = g.Cb ? g.Cb + cOff : nullptr;
  const float* R = g.R ? g.R + cOff : nullptr;
  const int cn  = lane & 15;
  const int cm0 = (lane >> 4) * 8;   // C/D layout: VGPR r -> M = cm0 + r
#pragma unroll
  for (int tm = 0; tm < 4; ++tm)
#pragma unroll
    for (int tn = 0; tn < 2; ++tn) {
      const int gn = n0 + wn + tn * 16 + cn;
      if (gn >= g.N) continue;
#pragma unroll
      for (int r = 0; r < 8; ++r) {
        const int gm = m0 + wm + tm * 16 + cm0 + r;
        const long long idx = (long long)gm * g.ldc + gn;
        float v = g.alpha * acc[tm][tn][r];
        if (R)  v += R[idx];
        if (Cf) Cf[idx] = v;
        if (Cb) Cb[idx] = (__bf16)v;
      }
    }
}

// ---------------------------------------------------------------------------
// LayerNorm over DIM=512, one wave32 per row, optional bf16 and/or f32 output
// ---------------------------------------------------------------------------
__global__ __launch_bounds__(256)
void ln_kernel(const float* __restrict__ x, const float* __restrict__ sc,
               const float* __restrict__ bi, __bf16* outb, float* outf, int rows) {
  const int row  = blockIdx.x * 8 + (threadIdx.x >> 5);
  const int lane = threadIdx.x & 31;
  if (row >= rows) return;
  const float* xr = x + (long long)row * DIM;
  float v[16]; float s = 0.f;
#pragma unroll
  for (int i = 0; i < 16; ++i) { v[i] = xr[lane + i * 32]; s += v[i]; }
#pragma unroll
  for (int o = 16; o; o >>= 1) s += __shfl_xor(s, o, 32);
  const float mean = s * (1.0f / DIM);
  float var = 0.f;
#pragma unroll
  for (int i = 0; i < 16; ++i) { float d = v[i] - mean; var += d * d; }
#pragma unroll
  for (int o = 16; o; o >>= 1) var += __shfl_xor(var, o, 32);
  const float rs = rsqrtf(var * (1.0f / DIM) + 1e-5f);
#pragma unroll
  for (int i = 0; i < 16; ++i) {
    const int c = lane + i * 32;
    const float y = (v[i] - mean) * rs * sc[c] + bi[c];
    if (outb) outb[(long long)row * DIM + c] = (__bf16)y;
    if (outf) outf[(long long)row * DIM + c] = y;
  }
}

__global__ void cvt_bf16_kernel(const float* __restrict__ in,
                                __bf16* __restrict__ out, long long n) {
  long long i = (long long)blockIdx.x * blockDim.x + threadIdx.x;
  const long long stride = (long long)gridDim.x * blockDim.x;
  for (; i < n; i += stride) out[i] = (__bf16)in[i];
}

__global__ void gelu_bf16_kernel(const float* __restrict__ in,
                                 __bf16* __restrict__ out, long long n) {
  long long i = (long long)blockIdx.x * blockDim.x + threadIdx.x;
  const long long stride = (long long)gridDim.x * blockDim.x;
  for (; i < n; i += stride) {
    const float x = in[i];
    const float t = tanhf(0.7978845608028654f * (x + 0.044715f * x * x * x));
    out[i] = (__bf16)(0.5f * x * (1.0f + t));
  }
}

// causal softmax over SEQ=512 logits, one wave per (bh,i) row -> bf16 probs
__global__ __launch_bounds__(256)
void softmax_causal_kernel(const float* __restrict__ S, __bf16* __restrict__ P,
                           int rows) {
  const int row  = blockIdx.x * 8 + (threadIdx.x >> 5);
  const int lane = threadIdx.x & 31;
  if (row >= rows) return;
  const int i = row & (SEQ - 1);
  const float* sr = S + (long long)row * SEQ;
  float v[16]; float mx = -1e30f;
#pragma unroll
  for (int t = 0; t < 16; ++t) {
    const int j = lane + t * 32;
    v[t] = (j <= i) ? sr[j] : -1e30f;
    mx = fmaxf(mx, v[t]);
  }
#pragma unroll
  for (int o = 16; o; o >>= 1) mx = fmaxf(mx, __shfl_xor(mx, o, 32));
  float se = 0.f;
#pragma unroll
  for (int t = 0; t < 16; ++t) {
    v[t] = (lane + t * 32 <= i) ? __expf(v[t] - mx) : 0.f;
    se += v[t];
  }
#pragma unroll
  for (int o = 16; o; o >>= 1) se += __shfl_xor(se, o, 32);
  const float inv = 1.0f / se;
#pragma unroll
  for (int t = 0; t < 16; ++t)
    P[(long long)row * SEQ + lane + t * 32] = (__bf16)(v[t] * inv);
}

// softmax over [TOPK mem logits | SEQ causal local logits]; writes a_mem (f32)
// and local probs (bf16)
__global__ __launch_bounds__(256)
void softmax_knn_kernel(const float* __restrict__ S, const float* __restrict__ tsim,
                        float* __restrict__ amem, __bf16* __restrict__ P, int rows) {
  const int row  = blockIdx.x * 8 + (threadIdx.x >> 5);
  const int lane = threadIdx.x & 31;
  if (row >= rows) return;
  const int i = row & (SEQ - 1);
  const float* sr = S + (long long)row * SEQ;
  const float mem = tsim[(long long)row * TOPK + lane];  // one per lane (TOPK==32)
  float v[16]; float mx = mem;
#pragma unroll
  for (int t = 0; t < 16; ++t) {
    const int j = lane + t * 32;
    v[t] = (j <= i) ? sr[j] : -1e30f;
    mx = fmaxf(mx, v[t]);
  }
#pragma unroll
  for (int o = 16; o; o >>= 1) mx = fmaxf(mx, __shfl_xor(mx, o, 32));
  const float em = __expf(mem - mx);
  float se = em;
#pragma unroll
  for (int t = 0; t < 16; ++t) {
    v[t] = (lane + t * 32 <= i) ? __expf(v[t] - mx) : 0.f;
    se += v[t];
  }
#pragma unroll
  for (int o = 16; o; o >>= 1) se += __shfl_xor(se, o, 32);
  const float inv = 1.0f / se;
  amem[(long long)row * TOPK + lane] = em * inv;
#pragma unroll
  for (int t = 0; t < 16; ++t)
    P[(long long)row * SEQ + lane + t * 32] = (__bf16)(v[t] * inv);
}

// top-32 of 4096: one wave per row, row cached in LDS, 32 x argmax sweeps
__global__ __launch_bounds__(128)
void topk32_kernel(const float* __restrict__ sim, float* __restrict__ tsim,
                   int* __restrict__ tidx, int rowBase) {
  __shared__ float buf[4][MEMN];
  const int wv = threadIdx.x >> 5, lane = threadIdx.x & 31;
  const int row = blockIdx.x * 4 + wv;
  const float* src = sim + (long long)row * MEMN;
  float* b = buf[wv];
  for (int j = lane; j < MEMN; j += 32) b[j] = src[j];
  const long long orow = (long long)(rowBase + row);
  for (int t = 0; t < TOPK; ++t) {
    float mx = -1e30f; int mi = 0;
    for (int j = lane; j < MEMN; j += 32) {
      const float v = b[j];
      if (v > mx) { mx = v; mi = j; }
    }
#pragma unroll
    for (int o = 16; o; o >>= 1) {
      const float om = __shfl_xor(mx, o, 32);
      const int   oi = __shfl_xor(mi, o, 32);
      if (om > mx || (om == mx && oi < mi)) { mx = om; mi = oi; }
    }
    if (lane == 0) {
      tsim[orow * TOPK + t] = mx;
      tidx[orow * TOPK + t] = mi;
      b[mi] = -1e30f;
    }
  }
}

// out[b,i,h*64+d] += sum_k a_mem[k] * mem_v[b, idx[k], d]; one wave per (bh,i)
__global__ __launch_bounds__(256)
void mem_gather_add_kernel(const float* __restrict__ amem,
                           const int* __restrict__ tidx,
                           const __bf16* __restrict__ mv,
                           float* __restrict__ out, int rows) {
  const int row  = blockIdx.x * 8 + (threadIdx.x >> 5);
  const int lane = threadIdx.x & 31;
  if (row >= rows) return;
  const int bh = row >> 9, i = row & (SEQ - 1);
  const int b = bh >> 3, h = bh & 7;
  const float* am = amem + (long long)row * TOPK;
  const int*   ix = tidx + (long long)row * TOPK;
  const __bf16* page = mv + (long long)b * MEMN * DH;
  float a0 = 0.f, a1 = 0.f;
  for (int k = 0; k < TOPK; ++k) {
    const float w = am[k];
    const __bf16* p = page + (long long)ix[k] * DH;
    a0 += w * (float)p[lane];
    a1 += w * (float)p[lane + 32];
  }
  float* o = out + ((long long)(b * SEQ + i) * DIM + h * DH);
  o[lane]      += a0;
  o[lane + 32] += a1;
}

// ---------------------------------------------------------------------------
// Host side
// ---------------------------------------------------------------------------
static void launch_gemm(hipStream_t s, const __bf16* A, const __bf16* B, float* Cf,
                        __bf16* Cb, const float* R, int M, int N, int K, int lda,
                        int ldb, int ldc, float alpha, int transB, int batch,
                        int shA, long long sAo, long long sAi,
                        int shB, long long sBo, long long sBi,
                        int shC, long long sCo, long long sCi, int zBase) {
  GemmArgs g;
  g.A = A; g.B = B; g.Cf = Cf; g.Cb = Cb; g.R = R;
  g.M = M; g.N = N; g.K = K; g.lda = lda; g.ldb = ldb; g.ldc = ldc;
  g.zShiftA = shA; g.zShiftB = shB; g.zShiftC = shC; g.zBase = zBase;
  g.sAo = sAo; g.sAi = sAi; g.sBo = sBo; g.sBi = sBi; g.sCo = sCo; g.sCi = sCi;
  g.alpha = alpha;
  dim3 grid((N + 127) / 128, (M + 127) / 128, batch);
  if (transB)
    gemm_bf16_wmma<1><<<grid, dim3(256), 0, s>>>(g);
  else
    gemm_bf16_wmma<0><<<grid, dim3(256), 0, s>>>(g);
}

static void launch_cvt(hipStream_t s, const float* in, __bf16* out, long long n) {
  int blocks = (int)((n + 255) / 256); if (blocks > 8192) blocks = 8192;
  hipLaunchKernelGGL(cvt_bf16_kernel, dim3(blocks), dim3(256), 0, s, in, out, n);
}

extern "C" void kernel_launch(void* const* d_in, const int* in_sizes, int n_in,
                              void* d_out, int out_size, void* d_ws, size_t ws_size,
                              hipStream_t stream) {
  (void)in_sizes; (void)n_in; (void)out_size; (void)ws_size;
  const float* in_x  = (const float*)d_in[0];
  const float* in_Wq = (const float*)d_in[1];
  const float* in_Wk = (const float*)d_in[2];
  const float* in_Wv = (const float*)d_in[3];
  const float* in_Wo = (const float*)d_in[4];
  const float* ln1s  = (const float*)d_in[5];
  const float* ln1b  = (const float*)d_in[6];
  const float* ln2s  = (const float*)d_in[7];
  const float* ln2b  = (const float*)d_in[8];
  const float* in_W1 = (const float*)d_in[9];
  const float* in_W2 = (const float*)d_in[10];
  const float* lnfs  = (const float*)d_in[11];
  const float* lnfb  = (const float*)d_in[12];
  const float* in_mk = (const float*)d_in[13];
  const float* in_mv = (const float*)d_in[14];

  const long long ROWS = (long long)BATCH * SEQ;       // 4096
  const long long DD   = (long long)DIM * DIM;         // 262144
  const long long BH   = (long long)BATCH * HEADS;     // 64

  // workspace layout (256B aligned bump allocator)
  char* base = (char*)d_ws; size_t off = 0;
  auto alloc = [&](size_t bytes) -> void* {
    void* p = base + off; off = (off + bytes + 255) & ~(size_t)255; return p;
  };
  float*  xb     = (float*)alloc(ROWS * DIM * 4);
  __bf16* wq_bf  = (__bf16*)alloc(DEPTH * DD * 2);
  __bf16* wk_bf  = (__bf16*)alloc(DEPTH * DD * 2);
  __bf16* wv_bf  = (__bf16*)alloc(DEPTH * DD * 2);
  __bf16* wo_bf  = (__bf16*)alloc(DEPTH * DD * 2);
  __bf16* w1_bf  = (__bf16*)alloc((long long)DEPTH * DIM * FFD * 2);
  __bf16* w2_bf  = (__bf16*)alloc((long long)DEPTH * FFD * DIM * 2);
  __bf16* mk_bf  = (__bf16*)alloc(2LL * BATCH * MEMN * DH * 2);
  __bf16* mv_bf  = (__bf16*)alloc(2LL * BATCH * MEMN * DH * 2);
  __bf16* h_bf   = (__bf16*)alloc(ROWS * DIM * 2);
  __bf16* q_bf   = (__bf16*)alloc(ROWS * DIM * 2);
  __bf16* k_bf   = (__bf16*)alloc(ROWS * DIM * 2);
  __bf16* v_bf   = (__bf16*)alloc(ROWS * DIM * 2);
  float*  scores = (float*)alloc(BH * SEQ * SEQ * 4);
  __bf16* P_bf   = (__bf16*)alloc(BH * SEQ * SEQ * 2);
  float*  attnout= (float*)alloc(ROWS * DIM * 4);
  __bf16* ao_bf  = (__bf16*)alloc(ROWS * DIM * 2);
  float*  simall = (float*)alloc(8LL * SEQ * MEMN * 4);  // also FF1 f32 scratch
  float*  tsim   = (float*)alloc(BH * SEQ * TOPK * 4);
  int*    tix    = (int*)alloc(BH * SEQ * TOPK * 4);
  float*  amem   = (float*)alloc(BH * SEQ * TOPK * 4);
  __bf16* g_bf   = (__bf16*)alloc(ROWS * FFD * 2);

  // one-time conversions + x copy
  launch_cvt(stream, in_Wq, wq_bf, DEPTH * DD);
  launch_cvt(stream, in_Wk, wk_bf, DEPTH * DD);
  launch_cvt(stream, in_Wv, wv_bf, DEPTH * DD);
  launch_cvt(stream, in_Wo, wo_bf, DEPTH * DD);
  launch_cvt(stream, in_W1, w1_bf, (long long)DEPTH * DIM * FFD);
  launch_cvt(stream, in_W2, w2_bf, (long long)DEPTH * FFD * DIM);
  launch_cvt(stream, in_mk, mk_bf, 2LL * BATCH * MEMN * DH);
  launch_cvt(stream, in_mv, mv_bf, 2LL * BATCH * MEMN * DH);
  hipMemcpyAsync(xb, in_x, ROWS * DIM * 4, hipMemcpyDeviceToDevice, stream);

  const float scale = 0.125f;  // DH^-0.5
  const long long sBH = (long long)SEQ * DIM;  // per-batch activation stride

  for (int l = 0; l < DEPTH; ++l) {
    const int knn = (l == 3 || l == 4);
    const int mi  = (l == 3) ? 0 : 1;

    // h = LN1(x) -> bf16
    hipLaunchKernelGGL(ln_kernel, dim3(512), dim3(256), 0, stream,
                       xb, ln1s + l * DIM, ln1b + l * DIM, h_bf, (float*)nullptr,
                       (int)ROWS);
    // q,k,v projections (bf16 out)
    launch_gemm(stream, h_bf, wq_bf + (long long)l * DD, nullptr, q_bf, nullptr,
                (int)ROWS, DIM, DIM, DIM, DIM, DIM, 1.0f, 0, 1,
                0, 0, 0, 0, 0, 0, 0, 0, 0, 0);
    launch_gemm(stream, h_bf, wk_bf + (long long)l * DD, nullptr, k_bf, nullptr,
                (int)ROWS, DIM, DIM, DIM, DIM, DIM, 1.0f, 0, 1,
                0, 0, 0, 0, 0, 0, 0, 0, 0, 0);
    launch_gemm(stream, h_bf, wv_bf + (long long)l * DD, nullptr, v_bf, nullptr,
                (int)ROWS, DIM, DIM, DIM, DIM, DIM, 1.0f, 0, 1,
                0, 0, 0, 0, 0, 0, 0, 0, 0, 0);

    // scores[bh] = scale * Q_h K_h^T  (z=(b,h): shift=3, strides sBH / DH)
    launch_gemm(stream, q_bf, k_bf, scores, nullptr, nullptr,
                SEQ, SEQ, DH, DIM, DIM, SEQ, scale, 1, (int)BH,
                3, sBH, DH, 3, sBH, DH,
                0, (long long)SEQ * SEQ, 0, 0);

    if (knn) {
      const __bf16* mkp = mk_bf + (long long)mi * BATCH * MEMN * DH;
      for (int c = 0; c < 8; ++c) {  // 8 batch-heads per chunk
        launch_gemm(stream, q_bf, mkp, simall, nullptr, nullptr,
                    SEQ, MEMN, DH, DIM, DH, MEMN, scale, 1, 8,
                    3, sBH, DH, 3, (long long)MEMN * DH, 0,
                    0, (long long)SEQ * MEMN, 0, c * 8);
        hipLaunchKernelGGL(topk32_kernel, dim3((8 * SEQ) / 4), dim3(128), 0,
                           stream, simall, tsim, tix, c * 8 * SEQ);
      }
      hipLaunchKernelGGL(softmax_knn_kernel, dim3((int)(BH * SEQ / 8)), dim3(256),
                         0, stream, scores, tsim, amem, P_bf, (int)(BH * SEQ));
    } else {
      hipLaunchKernelGGL(softmax_causal_kernel, dim3((int)(BH * SEQ / 8)),
                         dim3(256), 0, stream, scores, P_bf, (int)(BH * SEQ));
    }

    // attnout[b,i,h*64+d] = P[bh] @ V_h   (M=512, N=64, K=512)
    launch_gemm(stream, P_bf, v_bf, attnout, nullptr, nullptr,
                SEQ, DH, SEQ, SEQ, DIM, DIM, 1.0f, 0, (int)BH,
                0, (long long)SEQ * SEQ, 0, 3, sBH, DH,
                3, sBH, DH, 0);

    if (knn) {
      hipLaunchKernelGGL(mem_gather_add_kernel, dim3((int)(BH * SEQ / 8)),
                         dim3(256), 0, stream, amem, tix,
                         mv_bf + (long long)mi * BATCH * MEMN * DH, attnout,
                         (int)(BH * SEQ));
    }

    // x = x + attnout @ Wo
    launch_cvt(stream, attnout, ao_bf, ROWS * DIM);
    launch_gemm(stream, ao_bf, wo_bf + (long long)l * DD, xb, nullptr, xb,
                (int)ROWS, DIM, DIM, DIM, DIM, DIM, 1.0f, 0, 1,
                0, 0, 0, 0, 0, 0, 0, 0, 0, 0);

    // FF: x = x + gelu(LN2(x) @ W1) @ W2
    hipLaunchKernelGGL(ln_kernel, dim3(512), dim3(256), 0, stream,
                       xb, ln2s + l * DIM, ln2b + l * DIM, h_bf, (float*)nullptr,
                       (int)ROWS);
    launch_gemm(stream, h_bf, w1_bf + (long long)l * DIM * FFD, simall, nullptr,
                nullptr, (int)ROWS, FFD, DIM, DIM, FFD, FFD, 1.0f, 0, 1,
                0, 0, 0, 0, 0, 0, 0, 0, 0, 0);
    hipLaunchKernelGGL(gelu_bf16_kernel, dim3(8192), dim3(256), 0, stream,
                       simall, g_bf, ROWS * FFD);
    launch_gemm(stream, g_bf, w2_bf + (long long)l * FFD * DIM, xb, nullptr, xb,
                (int)ROWS, DIM, FFD, FFD, DIM, DIM, 1.0f, 0, 1,
                0, 0, 0, 0, 0, 0, 0, 0, 0, 0);
  }

  // final LN -> d_out (f32)
  hipLaunchKernelGGL(ln_kernel, dim3(512), dim3(256), 0, stream,
                     xb, lnfs, lnfb, (__bf16*)nullptr, (float*)d_out, (int)ROWS);
}
